// BRCA_Baseline_Simple_79104707658192
// MI455X (gfx1250) — matrix operand order repaired and checked by reading, not compiled, not run on hardware
//
#include <hip/hip_runtime.h>
#include <cstdint>

#define NROWS 8192
#define HID   256
#define NCLS  5
#define TOPK  10

typedef __attribute__((ext_vector_type(16))) _Float16 v16h;
typedef __attribute__((ext_vector_type(8)))  float    v8f;

union Frag  { v16h v; uint4 q[2]; };
union Half16 { _Float16 h[16]; uint4 q[2]; };

// A-fragment (16x32 f16): lane m=l&15, hi=l>>4 -> halves K = hi*8+{0..7}, 16+hi*8+{0..7}
__device__ __forceinline__ v16h frag_a(const _Float16* p, int hi) {
  Frag f;
  f.q[0] = *reinterpret_cast<const uint4*>(p + hi * 8);
  f.q[1] = *reinterpret_cast<const uint4*>(p + 16 + hi * 8);
  return f.v;
}
// B-fragment (32x16 f16): lane n=l&15, hi=l>>4 -> halves K = hi*8+{0..7}, 16+hi*8+{0..7}; p = row n (K-contiguous)
__device__ __forceinline__ v16h frag_b(const _Float16* p, int hi) {
  Frag f;
  f.q[0] = *reinterpret_cast<const uint4*>(p + hi * 8);
  f.q[1] = *reinterpret_cast<const uint4*>(p + 16 + hi * 8);
  return f.v;
}

__device__ __forceinline__ float wredx(float v) {
  #pragma unroll
  for (int o = 16; o > 0; o >>= 1) v += __shfl_xor(v, o, 32);
  return v;
}

// ---------------- GEMM: C[M,Nc] = act(A[M,K] @ B[K,Nc] + bias), f16 WMMA inside ----------------
__global__ __launch_bounds__(128) void k_gemm(
    const float* __restrict__ A, int lda, const float* __restrict__ B, int ldb,
    const float* __restrict__ bias, float* __restrict__ C, int ldc,
    int K, int act, int hasb) {
  __shared__ alignas(16) _Float16 As[64][40];   // 80B pitch -> 16B-aligned fragment loads
  __shared__ alignas(16) _Float16 Bs[64][40];   // stored [n][k]
  const int tid  = threadIdx.x;
  const int wave = tid >> 5, lane = tid & 31;
  const int m16  = lane & 15, hi = lane >> 4;
  const int n0 = blockIdx.x * 64, m0 = blockIdx.y * 64;

  // per-thread staging geometry (straight-line, no divergence)
  const int ar  = tid >> 1;            // A row 0..63
  const int ac0 = (tid & 1) * 16;      // A col 0 or 16
  const float* Arow = A + (size_t)(m0 + ar) * lda;
  const int bk  = tid >> 2;            // B k-row 0..31
  const int bn0 = (tid & 3) * 16;      // B col chunk

  v8f acc[4];
  #pragma unroll
  for (int nt = 0; nt < 4; ++nt) acc[nt] = (v8f){0.f,0.f,0.f,0.f,0.f,0.f,0.f,0.f};

  for (int k0 = 0; k0 < K; k0 += 32) {
    // ---- A tile 64x32: 16 independent scalar loads (lda not 16B-aligned), clamp+select tail
    {
      const float* Ap = Arow + k0;
      float av[16];
      #pragma unroll
      for (int u = 0; u < 16; ++u) {
        int idx = (k0 + ac0 + u < K) ? (ac0 + u) : 0;   // clamped addr, uniform EXEC
        av[u] = Ap[idx];
      }
      Half16 ah;
      #pragma unroll
      for (int u = 0; u < 16; ++u)
        ah.h[u] = (_Float16)((k0 + ac0 + u < K) ? av[u] : 0.f);
      *reinterpret_cast<uint4*>(&As[ar][ac0])     = ah.q[0];
      *reinterpret_cast<uint4*>(&As[ar][ac0 + 8]) = ah.q[1];
    }
    // ---- B tile 32x64: 4x b128 loads (ldb=256 aligned), transposed b16 stores
    {
      bool ok = (k0 + bk) < K;
      const float* Bp = B + (size_t)(k0 + (ok ? bk : 0)) * ldb + n0 + bn0;
      float4 b0 = *reinterpret_cast<const float4*>(Bp + 0);
      float4 b1 = *reinterpret_cast<const float4*>(Bp + 4);
      float4 b2 = *reinterpret_cast<const float4*>(Bp + 8);
      float4 b3 = *reinterpret_cast<const float4*>(Bp + 12);
      float vb[16] = {b0.x,b0.y,b0.z,b0.w, b1.x,b1.y,b1.z,b1.w,
                      b2.x,b2.y,b2.z,b2.w, b3.x,b3.y,b3.z,b3.w};
      #pragma unroll
      for (int u = 0; u < 16; ++u)
        Bs[bn0 + u][bk] = (_Float16)(ok ? vb[u] : 0.f);
    }
    __syncthreads();
    // issue all fragment loads first, then 4 back-to-back WMMAs (one DS wait, XDL pipelining)
    v16h av = frag_a(&As[wave * 16 + m16][0], hi);
    v16h bv[4];
    #pragma unroll
    for (int nt = 0; nt < 4; ++nt) bv[nt] = frag_b(&Bs[nt * 16 + m16][0], hi);
    #pragma unroll
    for (int nt = 0; nt < 4; ++nt)
      acc[nt] = __builtin_amdgcn_wmma_f32_16x16x32_f16(false, av, false, bv[nt],
                                                       (short)0, acc[nt], false, false);
    __syncthreads();
  }
  // hoist bias: one value per n-tile per lane (shared by all 8 rows of the tile)
  float bcol[4] = {0.f, 0.f, 0.f, 0.f};
  if (hasb) {
    #pragma unroll
    for (int nt = 0; nt < 4; ++nt) bcol[nt] = bias[n0 + nt * 16 + m16];
  }
  #pragma unroll
  for (int nt = 0; nt < 4; ++nt) {
    int col = n0 + nt * 16 + m16;
    #pragma unroll
    for (int r = 0; r < 8; ++r) {
      int row = m0 + wave * 16 + hi * 8 + r;    // C layout: VGPR r -> M = r + hi*8
      float v = acc[nt][r] + bcol[nt];
      if (act) v = fmaxf(v, 0.f);
      C[(size_t)row * ldc + col] = v;
    }
  }
}

// ---------------- BatchNorm stats / apply ----------------
__global__ __launch_bounds__(256) void k_bnstats(const float* __restrict__ X,
                                                 float* __restrict__ cs, float* __restrict__ cs2) {
  int col = threadIdx.x;
  float s = 0.f, s2 = 0.f;
  int r0 = blockIdx.x * 128;
  for (int rr = 0; rr < 128; ++rr) {
    float v = X[(size_t)(r0 + rr) * HID + col];
    s += v; s2 += v * v;
  }
  atomicAdd(&cs[col], s);
  atomicAdd(&cs2[col], s2);
}

__global__ __launch_bounds__(256) void k_bnapply(float* __restrict__ X,
    const float* __restrict__ cs, const float* __restrict__ cs2,
    const float* __restrict__ g, const float* __restrict__ b) {
  size_t i = (size_t)blockIdx.x * 256 + threadIdx.x;
  int col = (int)(i & (HID - 1));
  float mu  = cs[col]  * (1.f / NROWS);
  float var = cs2[col] * (1.f / NROWS) - mu * mu;
  X[i] = (X[i] - mu) * rsqrtf(var + 1e-5f) * g[col] + b[col];
}

// ---------------- row L2 normalize -> f16 ----------------
__global__ __launch_bounds__(256) void k_rownorm(const float* __restrict__ X,
                                                 _Float16* __restrict__ fn) {
  int wave = threadIdx.x >> 5, lane = threadIdx.x & 31;
  int row = blockIdx.x * 8 + wave;
  float ss = 0.f;
  #pragma unroll
  for (int u = 0; u < 8; ++u) {
    float v = X[(size_t)row * HID + lane + 32 * u];
    ss += v * v;
  }
  ss = wredx(ss);
  float rn = 1.f / fmaxf(sqrtf(ss), 1e-12f);
  #pragma unroll
  for (int u = 0; u < 8; ++u) {
    int col = lane + 32 * u;
    fn[(size_t)row * HID + col] = (_Float16)(X[(size_t)row * HID + col] * rn);
  }
}

// ---------------- fused cosine-sim GEMM + per-row top-10 (self excluded) ----------------
__global__ __launch_bounds__(128) void k_simtopk(const _Float16* __restrict__ fn,
    float* __restrict__ tvo, int* __restrict__ tio, float* __restrict__ rso) {
  __shared__ alignas(16) float simS[64][68];     // 272B pitch: 16B-aligned float4 scans
  __shared__ float mV[128][TOPK];
  __shared__ int   mI[128][TOPK];
  const int tid  = threadIdx.x;
  const int wave = tid >> 5, lane = tid & 31;
  const int m16  = lane & 15, hi = lane >> 4;
  const int i0 = blockIdx.x * 64;
  const int a_row = i0 + wave * 16 + m16;
  const int srow = tid >> 1;                     // scan: 2 threads per row
  const int shalf = (tid & 1) * 32;

  float tv[TOPK]; int ti[TOPK];
  #pragma unroll
  for (int q = 0; q < TOPK; ++q) { tv[q] = -3e38f; ti[q] = -1; }

  for (int j0 = 0; j0 < NROWS; j0 += 64) {
    if (j0 + 64 < NROWS)   // warm next B-block into cache (global_prefetch_b8)
      __builtin_prefetch(fn + (size_t)(j0 + 64 + (tid & 63)) * HID, 0, 0);

    v8f acc[4];
    #pragma unroll
    for (int nt = 0; nt < 4; ++nt) acc[nt] = (v8f){0.f,0.f,0.f,0.f,0.f,0.f,0.f,0.f};

    #pragma unroll 2
    for (int k0 = 0; k0 < HID; k0 += 32) {
      // all fragment loads in flight, then 4 back-to-back WMMAs
      v16h av = frag_a(fn + (size_t)a_row * HID + k0, hi);
      v16h bv[4];
      #pragma unroll
      for (int nt = 0; nt < 4; ++nt)
        bv[nt] = frag_b(fn + (size_t)(j0 + nt * 16 + m16) * HID + k0, hi); // B col n = fn row n
      #pragma unroll
      for (int nt = 0; nt < 4; ++nt)
        acc[nt] = __builtin_amdgcn_wmma_f32_16x16x32_f16(false, av, false, bv[nt],
                                                         (short)0, acc[nt], false, false);
    }
    #pragma unroll
    for (int nt = 0; nt < 4; ++nt)
      #pragma unroll
      for (int r = 0; r < 8; ++r)
        simS[wave * 16 + hi * 8 + r][nt * 16 + m16] = acc[nt][r];
    __syncthreads();

    {   // all 128 threads scan: thread pair (2r,2r+1) covers row r's 64 cols
      int row = i0 + srow;
      #pragma unroll
      for (int u = 0; u < 8; ++u) {
        float4 v4 = *reinterpret_cast<const float4*>(&simS[srow][shalf + 4 * u]);
        float vv[4] = {v4.x, v4.y, v4.z, v4.w};
        float gmax = fmaxf(fmaxf(vv[0], vv[1]), fmaxf(vv[2], vv[3]));
        if (gmax > tv[0]) {            // group pre-check: skip insert logic in common case
          #pragma unroll
          for (int e = 0; e < 4; ++e) {
            int j = j0 + shalf + 4 * u + e;
            float v = vv[e];
            if (j != row && v > tv[0]) {
              int p = 0;
              #pragma unroll
              for (int q = 1; q < TOPK; ++q)
                if (v > tv[q]) { tv[q-1] = tv[q]; ti[q-1] = ti[q]; p = q; }
              tv[p] = v; ti[p] = j;
            }
          }
        }
      }
    }
    __syncthreads();
  }

  // merge the two half-lists per row, then emit
  #pragma unroll
  for (int q = 0; q < TOPK; ++q) { mV[tid][q] = tv[q]; mI[tid][q] = ti[q]; }
  __syncthreads();
  if (tid < 64) {
    int row = i0 + tid;
    float fv[TOPK]; int fi[TOPK];
    #pragma unroll
    for (int q = 0; q < TOPK; ++q) { fv[q] = mV[2 * tid][q]; fi[q] = mI[2 * tid][q]; }
    #pragma unroll
    for (int q = 0; q < TOPK; ++q) {
      float v = mV[2 * tid + 1][q]; int j = mI[2 * tid + 1][q];
      if (v > fv[0]) {
        int p = 0;
        #pragma unroll
        for (int qq = 1; qq < TOPK; ++qq)
          if (v > fv[qq]) { fv[qq-1] = fv[qq]; fi[qq-1] = fi[qq]; p = qq; }
        fv[p] = v; fi[p] = j;
      }
    }
    float s = 1.0f;                     // identity weight
    #pragma unroll
    for (int q = 0; q < TOPK; ++q) {
      if (fi[q] >= 0) s += fabsf(fv[q]);
      tvo[(size_t)row * TOPK + q] = fv[q];
      tio[(size_t)row * TOPK + q] = fi[q];
    }
    rso[row] = s;
  }
}

// ---------------- SpMM: Y = leaky( (X[row] + sum v_k X[idx_k]) / rs + bias ) ----------------
__global__ __launch_bounds__(256) void k_spmm(const float* __restrict__ X,
    const float* __restrict__ tv, const int* __restrict__ ti, const float* __restrict__ rs,
    const float* __restrict__ bias, float* __restrict__ Y) {
  int wave = threadIdx.x >> 5, lane = threadIdx.x & 31;
  int row = blockIdx.x * 8 + wave;
  float inv = 1.f / rs[row];
  float acc[8];
  #pragma unroll
  for (int u = 0; u < 8; ++u) acc[u] = X[(size_t)row * HID + lane + 32 * u];
  #pragma unroll
  for (int k = 0; k < TOPK; ++k) {
    int idx = ti[(size_t)row * TOPK + k];
    float w = tv[(size_t)row * TOPK + k];
    if (idx >= 0) {
      #pragma unroll
      for (int u = 0; u < 8; ++u) acc[u] += w * X[(size_t)idx * HID + lane + 32 * u];
    }
  }
  #pragma unroll
  for (int u = 0; u < 8; ++u) {
    int col = lane + 32 * u;
    float v = acc[u] * inv + bias[col];
    Y[(size_t)row * HID + col] = (v >= 0.f) ? v : 0.25f * v;
  }
}

// ---------------- classifier head: logits = H @ W[256,5] + b ----------------
__global__ __launch_bounds__(256) void k_logits(const float* __restrict__ H,
    const float* __restrict__ W, const float* __restrict__ b, float* __restrict__ out) {
  int wave = threadIdx.x >> 5, lane = threadIdx.x & 31;
  int row = blockIdx.x * 8 + wave;
  float p[NCLS] = {0.f, 0.f, 0.f, 0.f, 0.f};
  #pragma unroll
  for (int u = 0; u < 8; ++u) {
    int col = lane + 32 * u;
    float x = H[(size_t)row * HID + col];
    #pragma unroll
    for (int c = 0; c < NCLS; ++c) p[c] += x * W[col * NCLS + c];
  }
  #pragma unroll
  for (int c = 0; c < NCLS; ++c) p[c] = wredx(p[c]);
  if (lane == 0)
    #pragma unroll
    for (int c = 0; c < NCLS; ++c) out[(size_t)row * NCLS + c] = p[c] + b[c];
}

// ---------------- VCDN fusion: softmax(attn)-weighted sigmoid + 2-layer MLP ----------------
__global__ __launch_bounds__(256) void k_fuse(const float* __restrict__ lg,
    const float* __restrict__ attn, const float* __restrict__ f1W, const float* __restrict__ f1b,
    const float* __restrict__ f2W, const float* __restrict__ f2b, float* __restrict__ out) {
  int row = blockIdx.x * 256 + threadIdx.x;
  float a0 = attn[0], a1 = attn[1], a2 = attn[2];
  float mx = fmaxf(a0, fmaxf(a1, a2));
  float e0 = expf(a0 - mx), e1 = expf(a1 - mx), e2 = expf(a2 - mx);
  float es = e0 + e1 + e2;
  float w[3] = {e0 / es, e1 / es, e2 / es};
  float fused[NCLS];
  #pragma unroll
  for (int c = 0; c < NCLS; ++c) {
    float s = 0.f;
    #pragma unroll
    for (int i = 0; i < 3; ++i) {
      float x = lg[(size_t)i * NROWS * NCLS + (size_t)row * NCLS + c];
      s += w[i] * (1.f / (1.f + expf(-x)));
    }
    fused[c] = s;
  }
  float acc[NCLS];
  #pragma unroll
  for (int c = 0; c < NCLS; ++c) acc[c] = f2b[c];
  for (int kk = 0; kk < 128; ++kk) {
    float s = f1b[kk];
    #pragma unroll
    for (int c = 0; c < NCLS; ++c) s += fused[c] * f1W[c * 128 + kk];
    float t = (s >= 0.f) ? s : 0.25f * s;
    #pragma unroll
    for (int c = 0; c < NCLS; ++c) acc[c] += t * f2W[kk * NCLS + c];
  }
  #pragma unroll
  for (int c = 0; c < NCLS; ++c) out[(size_t)row * NCLS + c] = acc[c];
}

extern "C" void kernel_launch(void* const* d_in, const int* in_sizes, int n_in,
                              void* d_out, int out_size, void* d_ws, size_t ws_size,
                              hipStream_t stream) {
  (void)n_in; (void)out_size; (void)ws_size;
  const float* x[3]    = {(const float*)d_in[0], (const float*)d_in[1], (const float*)d_in[2]};
  const float* encW[3] = {(const float*)d_in[3], (const float*)d_in[4], (const float*)d_in[5]};
  const float* enc_b = (const float*)d_in[6];
  const float* bn_g  = (const float*)d_in[7];
  const float* bn_b  = (const float*)d_in[8];
  const float* gc1W  = (const float*)d_in[9];
  const float* gc1b  = (const float*)d_in[10];
  const float* gc2W  = (const float*)d_in[11];
  const float* gc2b  = (const float*)d_in[12];
  const float* clfW  = (const float*)d_in[13];
  const float* clfb  = (const float*)d_in[14];
  const float* attn  = (const float*)d_in[15];
  const float* f1W   = (const float*)d_in[16];
  const float* f1b   = (const float*)d_in[17];
  const float* f2W   = (const float*)d_in[18];
  const float* f2b   = (const float*)d_in[19];

  const size_t F = (size_t)NROWS * HID;
  char* p = (char*)d_ws;
  float*    feat   = (float*)p;     p += F * 4;
  float*    tmp    = (float*)p;     p += F * 4;
  float*    hbuf   = (float*)p;     p += F * 4;
  _Float16* fn16   = (_Float16*)p;  p += F * 2;
  float*    logits = (float*)p;     p += (size_t)3 * NROWS * NCLS * 4;
  float*    tvb    = (float*)p;     p += (size_t)NROWS * TOPK * 4;
  int*      tib    = (int*)p;       p += (size_t)NROWS * TOPK * 4;
  float*    rsb    = (float*)p;     p += (size_t)NROWS * 4;
  float*    colsum = (float*)p;     p += HID * 4;     // colsum, colsq contiguous
  /* colsq = colsum + HID */

  const dim3 gemm_grid(HID / 64, NROWS / 64);

  for (int i = 0; i < 3; ++i) {
    int K = in_sizes[i] / NROWS;   // 1000, 1000, 503
    hipMemsetAsync(colsum, 0, 2 * HID * sizeof(float), stream);

    // encoder: relu(x @ W + b)
    k_gemm<<<gemm_grid, 128, 0, stream>>>(x[i], K, encW[i], HID, enc_b + i * HID,
                                          feat, HID, K, /*act*/1, /*hasb*/1);
    // batchnorm (batch stats)
    k_bnstats<<<NROWS / 128, 256, 0, stream>>>(feat, colsum, colsum + HID);
    k_bnapply<<<(int)(F / 256), 256, 0, stream>>>(feat, colsum, colsum + HID,
                                                  bn_g + i * HID, bn_b + i * HID);
    // cosine features + fused sim-GEMM/top-10
    k_rownorm<<<NROWS / 8, 256, 0, stream>>>(feat, fn16);
    k_simtopk<<<NROWS / 64, 128, 0, stream>>>(fn16, tvb, tib, rsb);
    // GCN layer 1
    k_gemm<<<gemm_grid, 128, 0, stream>>>(feat, HID, gc1W + (size_t)i * HID * HID, HID,
                                          nullptr, tmp, HID, HID, 0, 0);
    k_spmm<<<NROWS / 8, 256, 0, stream>>>(tmp, tvb, tib, rsb, gc1b + i * HID, hbuf);
    // GCN layer 2
    k_gemm<<<gemm_grid, 128, 0, stream>>>(hbuf, HID, gc2W + (size_t)i * HID * HID, HID,
                                          nullptr, tmp, HID, HID, 0, 0);
    k_spmm<<<NROWS / 8, 256, 0, stream>>>(tmp, tvb, tib, rsb, gc2b + i * HID, feat);
    // classifier
    k_logits<<<NROWS / 8, 256, 0, stream>>>(feat, clfW + (size_t)i * HID * NCLS,
                                            clfb + i * NCLS, logits + (size_t)i * NROWS * NCLS);
  }
  k_fuse<<<NROWS / 256, 256, 0, stream>>>(logits, attn, f1W, f1b, f2W, f2b, (float*)d_out);
}
